// SanaVideoTransformerBlock_37022618091582
// MI455X (gfx1250) — matrix operand order repaired and checked by reading, not compiled, not run on hardware
//
#include <hip/hip_runtime.h>
#include <math.h>

// ---------------- problem constants (match reference) ----------------
#define DIMC   2240
#define HEADS  20
#define HDC    112
#define FRM    8
#define HH     16
#define WW     28
#define HWC    (HH*WW)        // 448
#define SS     (FRM*HH*WW)    // 3584
#define ENCC   300
#define HIDC   6720           // 3*DIM
#define EPS_LN 1e-6f
#define EPS_RMS 1e-5f

// ---------------- WMMA types ----------------
typedef __attribute__((ext_vector_type(16))) __bf16 v16bf;
typedef __attribute__((ext_vector_type(8)))  float  v8f;

union FragU { unsigned u[8]; v16bf v; };

__device__ __forceinline__ unsigned pack_bf2(float a, float b) {
    unsigned ua = __float_as_uint(a);
    ua = (ua + 0x7FFFu + ((ua >> 16) & 1u)) >> 16;
    unsigned ub = __float_as_uint(b);
    ub = (ub + 0x7FFFu + ((ub >> 16) & 1u)) >> 16;
    return (ua & 0xFFFFu) | (ub << 16);
}

// ---------------- generic WMMA GEMM (register-staged pipeline) ----------------
// C[M,N] = act( alpha * op(A)[M,K] @ op(B)[K,N] + bias + (accum ? C : 0) )
// op(A): transA ? A[k*lda+m] : A[m*lda+k]
// op(B): transB ? B[n*ldb+k] : B[k*ldb+n]
// batched over gridDim.z with element strides sA,sB,sC. act: 0=none, 1=silu
#define TM 128
#define TN 128
#define TKK 32
#define NTH 256

__global__ __launch_bounds__(NTH)
void wmma_gemm_kernel(const float* __restrict__ A, const float* __restrict__ B,
                      const float* __restrict__ bias, float* __restrict__ C,
                      int M, int N, int K, int lda, int ldb, int ldc,
                      int transA, int transB, float alpha, int accum, int act,
                      long long sA, long long sB, long long sC)
{
    __shared__ unsigned As[TM * 17];   // [row m][k-pair 0..15], pad 17
    __shared__ unsigned Bs[TN * 17];   // [col n][k-pair 0..15], pad 17

    long long bz = blockIdx.z;
    A += bz * sA; B += bz * sB; C += bz * sC;

    const int mTile = blockIdx.y * TM;
    const int nTile = blockIdx.x * TN;
    const int tid   = threadIdx.x;
    const int wid   = tid >> 5;
    const int lane  = tid & 31;
    const int hi    = lane >> 4;
    const int l16   = lane & 15;
    const int waveM = wid & 3;   // 4 groups of 32 rows
    const int waveN = wid >> 2;  // 2 groups of 64 cols

    v8f acc[2][4];
#pragma unroll
    for (int a = 0; a < 2; ++a)
#pragma unroll
        for (int b = 0; b < 4; ++b) acc[a][b] = v8f{};

    unsigned ra[8], rb[8];

    // load next k-tile global -> regs (bf16-packed), zero-padded OOB
    auto load_tiles = [&](int k0) {
#pragma unroll
        for (int j = 0; j < 8; ++j) {
            int i = tid + j * NTH;           // A entries: [0, 2048)
            int row = i >> 4, p = i & 15;
            int m = mTile + row, k = k0 + 2 * p;
            float x0 = 0.f, x1 = 0.f;
            if (m < M) {
                if (!transA) {
                    const float* ap = A + (long long)m * lda + k;
                    if (k     < K) x0 = ap[0];
                    if (k + 1 < K) x1 = ap[1];
                } else {
                    if (k     < K) x0 = A[(long long)k * lda + m];
                    if (k + 1 < K) x1 = A[(long long)(k + 1) * lda + m];
                }
            }
            ra[j] = pack_bf2(x0, x1);
        }
#pragma unroll
        for (int j = 0; j < 8; ++j) {
            int i = tid + j * NTH;           // B entries: [0, 2048)
            int p = i >> 7, n = i & 127;
            int nn = nTile + n, k = k0 + 2 * p;
            float x0 = 0.f, x1 = 0.f;
            if (nn < N) {
                if (!transB) {
                    if (k     < K) x0 = B[(long long)k * ldb + nn];
                    if (k + 1 < K) x1 = B[(long long)(k + 1) * ldb + nn];
                } else {
                    const float* bp = B + (long long)nn * ldb + k;
                    if (k     < K) x0 = bp[0];
                    if (k + 1 < K) x1 = bp[1];
                }
            }
            rb[j] = pack_bf2(x0, x1);
        }
    };

    auto store_tiles = [&]() {
#pragma unroll
        for (int j = 0; j < 8; ++j) {
            int i = tid + j * NTH;
            As[(i >> 4) * 17 + (i & 15)] = ra[j];
        }
#pragma unroll
        for (int j = 0; j < 8; ++j) {
            int i = tid + j * NTH;
            Bs[(i & 127) * 17 + (i >> 7)] = rb[j];
        }
    };

    load_tiles(0);

    for (int k0 = 0; k0 < K; k0 += TKK) {
        store_tiles();
        __syncthreads();

        // prefetch next tile into regs; overlaps with ds loads + WMMA below
        if (k0 + TKK < K) load_tiles(k0 + TKK);

        // ---- fragment loads (ISA 7.12.2 layouts) ----
        FragU aF[2], bF[4];
#pragma unroll
        for (int mf = 0; mf < 2; ++mf) {
            int r = waveM * 32 + mf * 16 + l16;
#pragma unroll
            for (int v = 0; v < 8; ++v) {
                int p = (v < 4) ? (v + 4 * hi) : (v + 4 + 4 * hi); // A pair idx
                aF[mf].u[v] = As[r * 17 + p];
            }
        }
#pragma unroll
        for (int nf = 0; nf < 4; ++nf) {
            int n = waveN * 64 + nf * 16 + l16;
#pragma unroll
            for (int v = 0; v < 8; ++v)
                bF[nf].u[v] = Bs[n * 17 + ((v < 4) ? (v + 4 * hi) : (v + 4 + 4 * hi))]; // B pair idx
        }

#pragma unroll
        for (int mf = 0; mf < 2; ++mf)
#pragma unroll
            for (int nf = 0; nf < 4; ++nf)
                acc[mf][nf] = __builtin_amdgcn_wmma_f32_16x16x32_bf16(
                    false, aF[mf].v, false, bF[nf].v,
                    (short)0, acc[mf][nf], false, false);

        __syncthreads();
    }

    // ---- epilogue ----
#pragma unroll
    for (int mf = 0; mf < 2; ++mf) {
#pragma unroll
        for (int nf = 0; nf < 4; ++nf) {
            int col = nTile + waveN * 64 + nf * 16 + l16;
            if (col >= N) continue;
            float bv = bias ? bias[col] : 0.f;
#pragma unroll
            for (int v = 0; v < 8; ++v) {
                int row = mTile + waveM * 32 + mf * 16 + v + 8 * hi;
                if (row >= M) continue;
                long long off = (long long)row * ldc + col;
                float r = acc[mf][nf][v] * alpha + bv;
                if (accum) r += C[off];
                if (act == 1) r = r / (1.f + __expf(-r));   // silu
                C[off] = r;
            }
        }
    }
}

// ---------------- reductions helpers ----------------
__device__ __forceinline__ float blk_sum(float v, float* sm) {
    int t = threadIdx.x;
    sm[t] = v; __syncthreads();
    for (int s = blockDim.x >> 1; s > 0; s >>= 1) {
        if (t < s) sm[t] += sm[t + s];
        __syncthreads();
    }
    float r = sm[0]; __syncthreads();
    return r;
}
__device__ __forceinline__ float blk_max(float v, float* sm) {
    int t = threadIdx.x;
    sm[t] = v; __syncthreads();
    for (int s = blockDim.x >> 1; s > 0; s >>= 1) {
        if (t < s) sm[t] = fmaxf(sm[t], sm[t + s]);
        __syncthreads();
    }
    float r = sm[0]; __syncthreads();
    return r;
}

// ---------------- elementwise / small kernels ----------------
__global__ void mods_kernel(const float* sst, const float* ts, float* mods) {
    int i = blockIdx.x * blockDim.x + threadIdx.x;
    if (i < 6 * DIMC) mods[i] = sst[i] + ts[i];
}

// out = LN(x)*(1+sc)+sh, rows x DIM
__global__ void ln_mod_kernel(const float* __restrict__ x,
                              const float* __restrict__ sh,
                              const float* __restrict__ sc,
                              float* __restrict__ out) {
    __shared__ float sm[256];
    long long base = (long long)blockIdx.x * DIMC;
    float s = 0.f, s2 = 0.f;
    for (int d = threadIdx.x; d < DIMC; d += 256) {
        float v = x[base + d];
        s += v; s2 += v * v;
    }
    float tot  = blk_sum(s,  sm);
    float tot2 = blk_sum(s2, sm);
    float mean = tot / DIMC;
    float var  = tot2 / DIMC - mean * mean;
    float inv  = rsqrtf(var + EPS_LN);
    for (int d = threadIdx.x; d < DIMC; d += 256)
        out[base + d] = (x[base + d] - mean) * inv * (1.f + sc[d]) + sh[d];
}

// x = rms(x)*w (over DIM), optional relu; in-place
__global__ void rms_kernel(float* __restrict__ x, const float* __restrict__ w, int relu) {
    __shared__ float sm[256];
    long long base = (long long)blockIdx.x * DIMC;
    float s2 = 0.f;
    for (int d = threadIdx.x; d < DIMC; d += 256) {
        float v = x[base + d];
        s2 += v * v;
    }
    float tot2 = blk_sum(s2, sm);
    float inv = rsqrtf(tot2 / DIMC + EPS_RMS);
    for (int d = threadIdx.x; d < DIMC; d += 256) {
        float v = x[base + d] * inv * w[d];
        if (relu) v = fmaxf(v, 0.f);
        x[base + d] = v;
    }
}

// in-place RoPE over (S, HEADS, HD)
__global__ void rope_kernel(float* __restrict__ x,
                            const float* __restrict__ cs,
                            const float* __restrict__ sn) {
    int i = blockIdx.x * blockDim.x + threadIdx.x;
    if (i >= SS * (DIMC / 2)) return;
    int s  = i / (DIMC / 2);
    int j  = i % (DIMC / 2);
    int d2 = 2 * j;
    int dd = d2 % HDC;
    long long b = (long long)s * DIMC + d2;
    float x1 = x[b], x2 = x[b + 1];
    float c  = cs[(long long)s * HDC + dd];
    float si = sn[(long long)s * HDC + dd + 1];
    x[b]     = x1 * c - x2 * si;
    x[b + 1] = x1 * si + x2 * c;
}

__global__ void colsum_kernel(const float* __restrict__ k, float* __restrict__ ksum) {
    __shared__ float sm[256];
    int c = blockIdx.x;
    float a = 0.f;
    for (int s = threadIdx.x; s < SS; s += 256) a += k[(long long)s * DIMC + c];
    a = blk_sum(a, sm);
    if (threadIdx.x == 0) ksum[c] = a;
}

// denom[s,h] = <ksum_h, q_{s,h}> + 1e-15 ; ksum staged in LDS via async copy
__global__ void denom_kernel(const float* __restrict__ q,
                             const float* __restrict__ ksum,
                             float* __restrict__ den) {
    __shared__ float ks[DIMC];
#if defined(__gfx1250__)
    // async global -> LDS copy, 16B per lane per issue (gfx1250 ASYNCcnt path)
    for (int idx = threadIdx.x; idx < DIMC / 4; idx += 256) {
        unsigned lds_addr = (unsigned)(unsigned long long)(&ks[idx * 4]);
        const float* gp = ksum + idx * 4;
        asm volatile("global_load_async_to_lds_b128 %0, %1, off"
                     :: "v"(lds_addr), "v"(gp) : "memory");
    }
    asm volatile("s_wait_asynccnt 0" ::: "memory");
#else
    for (int d = threadIdx.x; d < DIMC; d += 256) ks[d] = ksum[d];
#endif
    __syncthreads();
    int i = blockIdx.x * 256 + threadIdx.x;
    if (i >= SS * HEADS) return;
    int s = i / HEADS, h = i % HEADS;
    long long qb = (long long)s * DIMC + h * HDC;
    int kb = h * HDC;
    float acc = 0.f;
#pragma unroll 4
    for (int d = 0; d < HDC; ++d) acc += q[qb + d] * ks[kb + d];
    den[i] = acc + 1e-15f;
}

__global__ void divdenom_kernel(float* __restrict__ o, const float* __restrict__ den) {
    long long i = (long long)blockIdx.x * blockDim.x + threadIdx.x;
    if (i >= (long long)SS * DIMC) return;
    int s = (int)(i / DIMC);
    int h = (int)((i % DIMC) / HDC);
    o[i] /= den[s * HEADS + h];
}

// x = h + g[d]*a
__global__ void addmul_kernel(const float* __restrict__ h, const float* __restrict__ g,
                              const float* __restrict__ a, float* __restrict__ x) {
    long long i = (long long)blockIdx.x * blockDim.x + threadIdx.x;
    if (i >= (long long)SS * DIMC) return;
    int d = (int)(i % DIMC);
    x[i] = h[i] + g[d] * a[i];
}

__global__ void addinplace_kernel(float* __restrict__ x, const float* __restrict__ c) {
    long long i = (long long)blockIdx.x * blockDim.x + threadIdx.x;
    if (i >= (long long)SS * DIMC) return;
    x[i] += c[i];
}

// softmax over rows of ENC
__global__ void softmax_kernel(float* __restrict__ p) {
    __shared__ float sm[256];
    float* row = p + (long long)blockIdx.x * ENCC;
    float m = -1e30f;
    for (int t = threadIdx.x; t < ENCC; t += 256) m = fmaxf(m, row[t]);
    m = blk_max(m, sm);
    float s = 0.f;
    for (int t = threadIdx.x; t < ENCC; t += 256) {
        float e = __expf(row[t] - m);
        row[t] = e; s += e;
    }
    s = blk_sum(s, sm);
    float inv = 1.f / s;
    for (int t = threadIdx.x; t < ENCC; t += 256) row[t] *= inv;
}

// depthwise 3x3 SAME on one frame: in/out (448, 2*HID)
__global__ void dwconv_kernel(const float* __restrict__ in, const float* __restrict__ w,
                              const float* __restrict__ b, float* __restrict__ out) {
    long long i = (long long)blockIdx.x * blockDim.x + threadIdx.x;
    const int C2 = 2 * HIDC;
    if (i >= (long long)HWC * C2) return;
    int c = (int)(i % C2);
    int r = (int)(i / C2);
    int y = r / WW, x = r % WW;
    float acc = b[c];
#pragma unroll
    for (int dy = -1; dy <= 1; ++dy) {
        int yy = y + dy;
        if (yy < 0 || yy >= HH) continue;
#pragma unroll
        for (int dx = -1; dx <= 1; ++dx) {
            int xx = x + dx;
            if (xx < 0 || xx >= WW) continue;
            acc += in[((long long)(yy * WW + xx)) * C2 + c] *
                   w[((long long)((dy + 1) * 3 + (dx + 1))) * C2 + c];
        }
    }
    out[i] = acc;
}

// o = a * silu(g) where y = [a | g] rows of 2*HID
__global__ void glu_kernel(const float* __restrict__ y, float* __restrict__ o) {
    long long i = (long long)blockIdx.x * blockDim.x + threadIdx.x;
    if (i >= (long long)HWC * HIDC) return;
    int r = (int)(i / HIDC), c = (int)(i % HIDC);
    float a = y[(long long)r * 2 * HIDC + c];
    float g = y[(long long)r * 2 * HIDC + HIDC + c];
    o[i] = a * (g / (1.f + __expf(-g)));
}

// out = x + g[d]*(ff + t)
__global__ void final_kernel(const float* __restrict__ x, const float* __restrict__ g,
                             const float* __restrict__ ff, const float* __restrict__ t,
                             float* __restrict__ out) {
    long long i = (long long)blockIdx.x * blockDim.x + threadIdx.x;
    if (i >= (long long)SS * DIMC) return;
    int d = (int)(i % DIMC);
    out[i] = x[i] + g[d] * (ff[i] + t[i]);
}

// ---------------- workspace arena ----------------
constexpr size_t ALN = 256;
constexpr size_t alup(size_t x) { return (x + ALN - 1) / ALN * ALN; }
constexpr size_t SD = (size_t)SS * DIMC;

constexpr size_t OFF_MODS = 0;
constexpr size_t OFF_NH   = alup(OFF_MODS + 6 * DIMC);
constexpr size_t OFF_Q    = alup(OFF_NH + SD);
constexpr size_t OFF_K    = alup(OFF_Q + SD);
constexpr size_t OFF_V    = alup(OFF_K + SD);
constexpr size_t OFF_K2   = alup(OFF_V + SD);
constexpr size_t OFF_V2   = alup(OFF_K2 + (size_t)ENCC * DIMC);
constexpr size_t OFF_KS   = alup(OFF_V2 + (size_t)ENCC * DIMC);
constexpr size_t OFF_DEN  = alup(OFF_KS + DIMC);
constexpr size_t OFF_SC   = alup(OFF_DEN + (size_t)SS * HEADS);
constexpr size_t OFF_LG   = alup(OFF_SC + (size_t)HEADS * HDC * HDC);
constexpr size_t OFF_F1   = alup(OFF_LG + (size_t)HEADS * SS * ENCC);
constexpr size_t OFF_F2   = alup(OFF_F1 + (size_t)HWC * 2 * HIDC);
constexpr size_t OFF_F3   = alup(OFF_F2 + (size_t)HWC * 2 * HIDC);
constexpr size_t WS_FLOATS = alup(OFF_F3 + (size_t)HWC * HIDC);

static inline int cdiv(long long a, long long b) { return (int)((a + b - 1) / b); }

static void gemm(hipStream_t st, const float* A, const float* B, const float* bias,
                 float* C, int M, int N, int K, int lda, int ldb, int ldc,
                 int tA, int tB, float alpha, int accum, int act,
                 int batch, long long sA, long long sB, long long sC) {
    dim3 grid(cdiv(N, TN), cdiv(M, TM), batch);
    wmma_gemm_kernel<<<grid, dim3(NTH), 0, st>>>(A, B, bias, C, M, N, K, lda, ldb, ldc,
                                                 tA, tB, alpha, accum, act, sA, sB, sC);
}

extern "C" void kernel_launch(void* const* d_in, const int* in_sizes, int n_in,
                              void* d_out, int out_size, void* d_ws, size_t ws_size,
                              hipStream_t stream) {
    (void)in_sizes; (void)n_in; (void)out_size;
    if (ws_size < WS_FLOATS * sizeof(float)) return;  // not enough scratch

    const float* hidden = (const float*)d_in[0];
    const float* enc    = (const float*)d_in[1];
    const float* tstep  = (const float*)d_in[2];
    const float* fcos   = (const float*)d_in[3];
    const float* fsin   = (const float*)d_in[4];
    const float* sst    = (const float*)d_in[5];
    const float* a1_qw = (const float*)d_in[6],  *a1_qb = (const float*)d_in[7];
    const float* a1_kw = (const float*)d_in[8],  *a1_kb = (const float*)d_in[9];
    const float* a1_vw = (const float*)d_in[10], *a1_vb = (const float*)d_in[11];
    const float* a1_nq = (const float*)d_in[12], *a1_nk = (const float*)d_in[13];
    const float* a1_ow = (const float*)d_in[14], *a1_ob = (const float*)d_in[15];
    const float* a2_qw = (const float*)d_in[16], *a2_qb = (const float*)d_in[17];
    const float* a2_kw = (const float*)d_in[18], *a2_kb = (const float*)d_in[19];
    const float* a2_vw = (const float*)d_in[20], *a2_vb = (const float*)d_in[21];
    const float* a2_nq = (const float*)d_in[22], *a2_nk = (const float*)d_in[23];
    const float* a2_ow = (const float*)d_in[24], *a2_ob = (const float*)d_in[25];
    const float* inv_w   = (const float*)d_in[26], *inv_b   = (const float*)d_in[27];
    const float* depth_w = (const float*)d_in[28], *depth_b = (const float*)d_in[29];
    const float* point_w = (const float*)d_in[30];
    const float* temp_w  = (const float*)d_in[31];

    float* ws   = (float*)d_ws;
    float* MODS = ws + OFF_MODS;
    float* NH   = ws + OFF_NH;
    float* Q    = ws + OFF_Q;
    float* Kb   = ws + OFF_K;
    float* V    = ws + OFF_V;
    float* K2   = ws + OFF_K2;
    float* V2   = ws + OFF_V2;
    float* KS   = ws + OFF_KS;
    float* DEN  = ws + OFF_DEN;
    float* SC   = ws + OFF_SC;
    float* LG   = ws + OFF_LG;
    float* F1   = ws + OFF_F1;
    float* F2   = ws + OFF_F2;
    float* F3   = ws + OFF_F3;
    float* out  = (float*)d_out;

    const long long nSD = (long long)SS * DIMC;
    const int eBlocks = cdiv(nSD, 256);

    // modulation vectors
    mods_kernel<<<cdiv(6 * DIMC, 256), 256, 0, stream>>>(sst, tstep, MODS);

    // nh = LN(hidden)*(1+sc_msa)+sh_msa
    ln_mod_kernel<<<SS, 256, 0, stream>>>(hidden, MODS + 0 * DIMC, MODS + 1 * DIMC, NH);

    // ---- linear attention (a1) ----
    gemm(stream, NH, a1_qw, a1_qb, Q, SS, DIMC, DIMC, DIMC, DIMC, DIMC, 0, 0, 1.f, 0, 0, 1, 0, 0, 0);
    rms_kernel<<<SS, 256, 0, stream>>>(Q, a1_nq, 1);
    gemm(stream, NH, a1_kw, a1_kb, Kb, SS, DIMC, DIMC, DIMC, DIMC, DIMC, 0, 0, 1.f, 0, 0, 1, 0, 0, 0);
    rms_kernel<<<SS, 256, 0, stream>>>(Kb, a1_nk, 1);
    gemm(stream, NH, a1_vw, a1_vb, V, SS, DIMC, DIMC, DIMC, DIMC, DIMC, 0, 0, 1.f, 0, 0, 1, 0, 0, 0);

    colsum_kernel<<<DIMC, 256, 0, stream>>>(Kb, KS);
    denom_kernel<<<cdiv((long long)SS * HEADS, 256), 256, 0, stream>>>(Q, KS, DEN);

    rope_kernel<<<cdiv((long long)SS * DIMC / 2, 256), 256, 0, stream>>>(Q, fcos, fsin);
    rope_kernel<<<cdiv((long long)SS * DIMC / 2, 256), 256, 0, stream>>>(Kb, fcos, fsin);

    // scores[h] = V_h^T (112xS) @ Kr_h (Sx112)
    gemm(stream, V, Kb, nullptr, SC, HDC, HDC, SS, DIMC, DIMC, HDC, 1, 0, 1.f, 0, 0,
         HEADS, HDC, HDC, (long long)HDC * HDC);
    // out[h] = Qr_h (Sx112) @ scores_h^T -> NH
    gemm(stream, Q, SC, nullptr, NH, SS, HDC, HDC, DIMC, HDC, DIMC, 0, 1, 1.f, 0, 0,
         HEADS, HDC, (long long)HDC * HDC, HDC);
    divdenom_kernel<<<eBlocks, 256, 0, stream>>>(NH, DEN);

    // o-projection -> Q, then x = hidden + g_msa * proj -> V
    gemm(stream, NH, a1_ow, a1_ob, Q, SS, DIMC, DIMC, DIMC, DIMC, DIMC, 0, 0, 1.f, 0, 0, 1, 0, 0, 0);
    addmul_kernel<<<eBlocks, 256, 0, stream>>>(hidden, MODS + 2 * DIMC, Q, V);

    // ---- cross attention (a2) ----
    gemm(stream, V, a2_qw, a2_qb, Q, SS, DIMC, DIMC, DIMC, DIMC, DIMC, 0, 0, 1.f, 0, 0, 1, 0, 0, 0);
    rms_kernel<<<SS, 256, 0, stream>>>(Q, a2_nq, 0);
    gemm(stream, enc, a2_kw, a2_kb, K2, ENCC, DIMC, DIMC, DIMC, DIMC, DIMC, 0, 0, 1.f, 0, 0, 1, 0, 0, 0);
    rms_kernel<<<ENCC, 256, 0, stream>>>(K2, a2_nk, 0);
    gemm(stream, enc, a2_vw, a2_vb, V2, ENCC, DIMC, DIMC, DIMC, DIMC, DIMC, 0, 0, 1.f, 0, 0, 1, 0, 0, 0);

    // logits[h] = Q_h (Sx112) @ K2_h^T (112xENC) / sqrt(112)
    gemm(stream, Q, K2, nullptr, LG, SS, ENCC, HDC, DIMC, DIMC, ENCC, 0, 1,
         0.094491118252306806f, 0, 0, HEADS, HDC, HDC, (long long)SS * ENCC);
    softmax_kernel<<<HEADS * SS, 256, 0, stream>>>(LG);
    // attn@V -> NH
    gemm(stream, LG, V2, nullptr, NH, SS, HDC, ENCC, ENCC, DIMC, DIMC, 0, 0, 1.f, 0, 0,
         HEADS, (long long)SS * ENCC, HDC, HDC);
    // o-projection -> Kb ; x += cross (in V)
    gemm(stream, NH, a2_ow, a2_ob, Kb, SS, DIMC, DIMC, DIMC, DIMC, DIMC, 0, 0, 1.f, 0, 0, 1, 0, 0, 0);
    addinplace_kernel<<<eBlocks, 256, 0, stream>>>(V, Kb);

    // ---- MLP ----
    // nh2 = LN(x)*(1+sc_mlp)+sh_mlp -> Q
    ln_mod_kernel<<<SS, 256, 0, stream>>>(V, MODS + 3 * DIMC, MODS + 4 * DIMC, Q);

    const int C2 = 2 * HIDC;
    for (int f = 0; f < FRM; ++f) {
        const float* nf = Q + (long long)f * HWC * DIMC;
        float* ffo = NH + (long long)f * HWC * DIMC;
        // inverted 1x1 conv + bias + silu
        gemm(stream, nf, inv_w, inv_b, F1, HWC, C2, DIMC, DIMC, C2, C2, 0, 0, 1.f, 0, 1, 1, 0, 0, 0);
        // depthwise 3x3 + bias
        dwconv_kernel<<<cdiv((long long)HWC * C2, 256), 256, 0, stream>>>(F1, depth_w, depth_b, F2);
        // GLU: a * silu(g)
        glu_kernel<<<cdiv((long long)HWC * HIDC, 256), 256, 0, stream>>>(F2, F3);
        // pointwise 1x1 conv
        gemm(stream, F3, point_w, nullptr, ffo, HWC, DIMC, HIDC, HIDC, DIMC, DIMC, 0, 0, 1.f, 0, 0, 1, 0, 0, 0);
    }

    // temporal conv (kernel 3 over frames, pad 1) -> Kb
    // center tap (w[1]) overwrites
    gemm(stream, NH, temp_w + (long long)DIMC * DIMC, nullptr, Kb,
         SS, DIMC, DIMC, DIMC, DIMC, DIMC, 0, 0, 1.f, 0, 0, 1, 0, 0, 0);
    // out frames 1..F-1 += in frames 0..F-2 @ w[0]
    gemm(stream, NH, temp_w, nullptr, Kb + (long long)HWC * DIMC,
         (FRM - 1) * HWC, DIMC, DIMC, DIMC, DIMC, DIMC, 0, 0, 1.f, 1, 0, 1, 0, 0, 0);
    // out frames 0..F-2 += in frames 1..F-1 @ w[2]
    gemm(stream, NH + (long long)HWC * DIMC, temp_w + 2LL * DIMC * DIMC, nullptr, Kb,
         (FRM - 1) * HWC, DIMC, DIMC, DIMC, DIMC, DIMC, 0, 0, 1.f, 1, 0, 1, 0, 0, 0);

    // out = x + g_mlp*(ffout + temporal)
    final_kernel<<<eBlocks, 256, 0, stream>>>(V, MODS + 5 * DIMC, NH, Kb, out);
}
